// GroupedQueryAttention_88399016886776
// MI455X (gfx1250) — compile-verified, matrix operand order checked by reading, not run on hardware
//
#include <hip/hip_runtime.h>
#include <hip/hip_bf16.h>

// ---------------------------------------------------------------- constants
#define BATCH     2
#define SEQ       2048
#define EMB       2048
#define NHEADS    32
#define KVH       8
#define HD        64
#define KVW       (KVH * HD)        // 512
#define BL        (BATCH * SEQ)     // 4096

typedef __attribute__((ext_vector_type(16))) __bf16 v16bf;
typedef __attribute__((ext_vector_type(8)))  float  v8f;

// ----------------------------------------------------- async global->LDS copy
// GLOBAL_LOAD_ASYNC_TO_LDS_B128 (ISA 15.18.3 op 98): per-lane 16B copy,
// tracked by ASYNCcnt, no VGPR round-trip.  Emitted via inline asm (the clang
// builtin prototype differs between toolchains).  Generic pointers to
// __shared__ carry the LDS byte offset in their low 32 bits on amdgcn.
__device__ __forceinline__ void async_copy_b128(__bf16* lds_dst, const __bf16* gsrc) {
    unsigned lds_off = (unsigned)(size_t)lds_dst;
    asm volatile("global_load_async_to_lds_b128 %0, %1, off"
                 :
                 : "v"(lds_off), "v"(gsrc)
                 : "memory");
}
__device__ __forceinline__ void wait_async_le4() {   // current tile done, next in flight
    asm volatile("s_wait_asynccnt 0x4" ::: "memory");
}
__device__ __forceinline__ void wait_async_0() {
    asm volatile("s_wait_asynccnt 0x0" ::: "memory");
}

// ---------------------------------------------------------------- helpers
__device__ __forceinline__ v8f vzero8() {
    v8f z;
#pragma unroll
    for (int i = 0; i < 8; ++i) z[i] = 0.0f;
    return z;
}

__device__ __forceinline__ v8f wmma_bf16(v16bf a, v16bf b, v8f c) {
    return __builtin_amdgcn_wmma_f32_16x16x32_bf16(
        false, a, false, b, (short)0, c, false, false);
}

// 16-bit A-layout fragment (ISA 7.12.2): lane r(&15) = row, VGPR v<4 -> K pairs
// 2v(+8*hi), v>=4 -> 16+2(v-4)(+8*hi).  K pairs contiguous -> merges to b128.
__device__ __forceinline__ v16bf load_frag(const __bf16* base, int n_stride, int lane) {
    int r  = lane & 15;
    int hi = (lane >> 4) & 1;
    const __bf16* p = base + (size_t)r * n_stride;
    v16bf a;
#pragma unroll
    for (int v = 0; v < 8; ++v) {
        int k0 = ((v < 4) ? (2 * v) : (8 + 2 * v)) + hi * 8;
        a[2 * v]     = p[k0];
        a[2 * v + 1] = p[k0 + 1];
    }
    return a;
}

// ---------------------------------------------------------------- f32 -> bf16
__global__ __launch_bounds__(256) void cvt_bf16_kernel(const float* __restrict__ s,
                                                       __bf16* __restrict__ d, long n) {
    long i      = (long)blockIdx.x * blockDim.x + threadIdx.x;
    long stride = (long)gridDim.x * blockDim.x;
    for (; i < n; i += stride) d[i] = (__bf16)s[i];
}

// f32 [K,N] -> bf16 transposed [N,K], LDS-tiled 32x32 so both sides coalesce
__global__ __launch_bounds__(256) void cvtT_bf16_kernel(const float* __restrict__ s,
                                                        __bf16* __restrict__ d,
                                                        int K, int N) {
    __shared__ __bf16 tile[32][33];
    int kb = blockIdx.y * 32, nb = blockIdx.x * 32;
    int tx = threadIdx.x & 31, ty = threadIdx.x >> 5;   // ty 0..7
#pragma unroll
    for (int j = 0; j < 4; ++j)
        tile[ty * 4 + j][tx] = (__bf16)s[(size_t)(kb + ty * 4 + j) * N + nb + tx];
    __syncthreads();
#pragma unroll
    for (int j = 0; j < 4; ++j)
        d[(size_t)(nb + ty * 4 + j) * K + kb + tx] = tile[tx][ty * 4 + j];
}

// ---------------------------------------------------------------- GEMM
// C[M,N] = A[M,K] * BT[N,K]^T + bias[N]   (both operands staged row-major)
// 128x128x32 tiles, 8 waves in 2x4 grid, wave = 64x32 = 4x2 WMMA accs.
// Double-buffered async global->LDS staging.
#define GBM 128
#define GBN 128
#define GBK 32
#define LDA 40   // 32+8 pad: 80B rows (16B aligned, conflict-free frag reads)

__global__ __launch_bounds__(256) void gemm_bf16_kernel(
    const __bf16* __restrict__ A, const __bf16* __restrict__ BT,
    const float* __restrict__ bias, int M, int N, int K,
    float* outF32, __bf16* outBF16, int bf16mode /*0:[M,N] 1:[b,kv,l,d] 2:[b,kv,d,l]*/,
    float* outKV) {

    __shared__ __bf16 As[2][GBM][LDA];
    __shared__ __bf16 Bs[2][GBN][LDA];

    const int t      = threadIdx.x;
    const int lane   = t & 31;
    const int wave   = t >> 5;
    const int wm     = wave >> 2;
    const int wn     = wave & 3;
    const int tile_m = blockIdx.y * GBM;
    const int tile_n = blockIdx.x * GBN;

    const int ar = t >> 1;              // 0..127: staging row
    const int ac = (t & 1) * 16;        // 0 / 16: staging col

    v8f acc[4][2];
#pragma unroll
    for (int i = 0; i < 4; ++i)
#pragma unroll
        for (int j = 0; j < 2; ++j) acc[i][j] = vzero8();

    auto stage = [&](int buf, int k0) {
        const __bf16* ag = A  + (size_t)(tile_m + ar) * K + k0 + ac;
        const __bf16* bg = BT + (size_t)(tile_n + ar) * K + k0 + ac;
        async_copy_b128(&As[buf][ar][ac],     ag);
        async_copy_b128(&As[buf][ar][ac + 8], ag + 8);
        async_copy_b128(&Bs[buf][ar][ac],     bg);
        async_copy_b128(&Bs[buf][ar][ac + 8], bg + 8);
    };

    const int nk = K / GBK;
    stage(0, 0);
    for (int ki = 0; ki < nk; ++ki) {
        const int buf = ki & 1;
        if (ki + 1 < nk) { stage(buf ^ 1, (ki + 1) * GBK); wait_async_le4(); }
        else             { wait_async_0(); }
        __syncthreads();

        v16bf af[4], bf[2];
#pragma unroll
        for (int i = 0; i < 4; ++i)
            af[i] = load_frag(&As[buf][wm * 64 + 16 * i][0], LDA, lane);
#pragma unroll
        for (int j = 0; j < 2; ++j)
            bf[j] = load_frag(&Bs[buf][wn * 32 + 16 * j][0], LDA, lane);
#pragma unroll
        for (int i = 0; i < 4; ++i)
#pragma unroll
            for (int j = 0; j < 2; ++j)
                acc[i][j] = wmma_bf16(af[i], bf[j], acc[i][j]);

        __syncthreads();   // protect buf before it is re-staged at ki+2
    }

    // epilogue: C layout VGPR i -> row i+8*(lane>=16), col = lane&15
    const int hi = lane >> 4;
    const int nn = lane & 15;
#pragma unroll
    for (int i = 0; i < 4; ++i)
#pragma unroll
        for (int j = 0; j < 2; ++j) {
            v8f c = acc[i][j];
#pragma unroll
            for (int ii = 0; ii < 8; ++ii) {
                int m = tile_m + wm * 64 + 16 * i + ii + 8 * hi;
                int n = tile_n + wn * 32 + 16 * j + nn;
                float val = c[ii] + bias[n];
                if (outF32) outF32[(size_t)m * N + n] = val;
                if (outBF16) {
                    if (bf16mode == 0) {
                        outBF16[(size_t)m * N + n] = (__bf16)val;
                    } else {
                        int bb = m / SEQ, l = m % SEQ;
                        int kvh = n >> 6, d = n & 63;
                        size_t idx = (bf16mode == 1)
                            ? (((size_t)(bb * KVH + kvh)) * SEQ + l) * HD + d
                            : (((size_t)(bb * KVH + kvh)) * HD + d) * SEQ + l;
                        outBF16[idx] = (__bf16)val;
                    }
                }
                if (outKV) { // f32 [B,Hk,L,D] for d_out
                    int bb = m / SEQ, l = m % SEQ;
                    int kvh = n >> 6, d = n & 63;
                    outKV[(((size_t)(bb * KVH + kvh)) * SEQ + l) * HD + d] = val;
                }
            }
        }
}

// ---------------------------------------------------------------- attention
// block = (b, h, 128 q-rows); 8 waves x 16 q-rows; flash over 64-key tiles.
// K staged from [b,kv,l,d], V staged from the d-major [b,kv,d,l] layout so
// both score- and PV-fragments are contiguous-pair LDS reads (ds_load_b128).
#define KLDS 72   // 64+8 pad: 144B rows (16B aligned)

__global__ __launch_bounds__(256) void attn_kernel(
    const __bf16* __restrict__ Qb,   // [B*L, E]
    const __bf16* __restrict__ Kt,   // [B, KVH, L, D]
    const __bf16* __restrict__ Vt,   // [B, KVH, D, L]
    __bf16* __restrict__ zb) {       // [B*L, E]

    __shared__ __bf16 Ks[2][64][KLDS];
    __shared__ __bf16 Vs[2][64][KLDS];
    __shared__ __bf16 Ps[8][16][KLDS];

    const int b    = blockIdx.z;
    const int h    = blockIdx.y;
    const int kv   = h & (KVH - 1);
    const int wave = threadIdx.x >> 5;
    const int lane = threadIdx.x & 31;
    const int hi   = lane >> 4;
    const int nn   = lane & 15;
    const int q0   = blockIdx.x * 128 + wave * 16;

    const __bf16* qbase = Qb + ((size_t)(b * SEQ + q0)) * EMB + h * HD;
    const v16bf qf0 = load_frag(qbase,      EMB, lane);
    const v16bf qf1 = load_frag(qbase + 32, EMB, lane);

    const __bf16* kbase = Kt + ((size_t)(b * KVH + kv) * SEQ) * HD;  // [l][d]
    const __bf16* vbase = Vt + ((size_t)(b * KVH + kv) * HD) * SEQ;  // [d][l]

    const int sr  = threadIdx.x >> 2;        // 0..63 staging row
    const int sc0 = (threadIdx.x & 3) * 16;  // 0..48 staging col

    auto stageKV = [&](int buf, int kb) {
        const __bf16* kg = kbase + (size_t)(kb + sr) * HD + sc0;   // row l, cols d
        const __bf16* vg = vbase + (size_t)sr * SEQ + kb + sc0;    // row d, cols l
        async_copy_b128(&Ks[buf][sr][sc0],     kg);
        async_copy_b128(&Ks[buf][sr][sc0 + 8], kg + 8);
        async_copy_b128(&Vs[buf][sr][sc0],     vg);
        async_copy_b128(&Vs[buf][sr][sc0 + 8], vg + 8);
    };

    v8f of[4];
#pragma unroll
    for (int j = 0; j < 4; ++j) of[j] = vzero8();
    float mrow[8], lrow[8];
#pragma unroll
    for (int i = 0; i < 8; ++i) { mrow[i] = -1e30f; lrow[i] = 0.0f; }

    const float sc = 0.125f;   // 1/sqrt(64)
    const int nit = SEQ / 64;

    stageKV(0, 0);
    for (int it = 0; it < nit; ++it) {
        const int buf = it & 1;
        if (it + 1 < nit) { stageKV(buf ^ 1, (it + 1) * 64); wait_async_le4(); }
        else              { wait_async_0(); }
        __syncthreads();

        // ---- S = Q K^T
        v8f sf[4];
#pragma unroll
        for (int j = 0; j < 4; ++j) {
            sf[j] = vzero8();
            v16bf bk0 = load_frag(&Ks[buf][16 * j][0],  KLDS, lane);
            v16bf bk1 = load_frag(&Ks[buf][16 * j][32], KLDS, lane);
            sf[j] = wmma_bf16(qf0, bk0, sf[j]);
            sf[j] = wmma_bf16(qf1, bk1, sf[j]);
        }

        // ---- online softmax (VGPR slot i <-> row i + 8*hi)
#pragma unroll
        for (int i = 0; i < 8; ++i) {
            float rm = fmaxf(fmaxf(sf[0][i], sf[1][i]), fmaxf(sf[2][i], sf[3][i]));
            rm = fmaxf(rm, __shfl_xor(rm, 1, 32));
            rm = fmaxf(rm, __shfl_xor(rm, 2, 32));
            rm = fmaxf(rm, __shfl_xor(rm, 4, 32));
            rm = fmaxf(rm, __shfl_xor(rm, 8, 32));
            float mn    = fmaxf(mrow[i], rm * sc);
            float alpha = __expf(mrow[i] - mn);
            float ps = 0.0f;
#pragma unroll
            for (int j = 0; j < 4; ++j) {
                float p = __expf(sf[j][i] * sc - mn);
                ps += p;
                Ps[wave][i + 8 * hi][16 * j + nn] = (__bf16)p;
            }
            ps += __shfl_xor(ps, 1, 32);
            ps += __shfl_xor(ps, 2, 32);
            ps += __shfl_xor(ps, 4, 32);
            ps += __shfl_xor(ps, 8, 32);
            lrow[i] = lrow[i] * alpha + ps;
            mrow[i] = mn;
#pragma unroll
            for (int j = 0; j < 4; ++j) of[j][i] *= alpha;
        }

        // ---- O += P V  (P re-read as A frags; V^T rows are d-major -> b128)
        v16bf pf0 = load_frag(&Ps[wave][0][0],  KLDS, lane);
        v16bf pf1 = load_frag(&Ps[wave][0][32], KLDS, lane);
#pragma unroll
        for (int jd = 0; jd < 4; ++jd) {
            v16bf bv0 = load_frag(&Vs[buf][16 * jd][0],  KLDS, lane);
            v16bf bv1 = load_frag(&Vs[buf][16 * jd][32], KLDS, lane);
            of[jd] = wmma_bf16(pf0, bv0, of[jd]);
            of[jd] = wmma_bf16(pf1, bv1, of[jd]);
        }
        __syncthreads();
    }

    // ---- normalize + write z (head h = output slot h; kv = h%8)
#pragma unroll
    for (int i = 0; i < 8; ++i) {
        float rl  = 1.0f / lrow[i];
        int   row = q0 + i + 8 * hi;
#pragma unroll
        for (int jd = 0; jd < 4; ++jd)
            zb[((size_t)(b * SEQ + row)) * EMB + h * HD + 16 * jd + nn] =
                (__bf16)(of[jd][i] * rl);
    }
}

// ---------------------------------------------------------------- launcher
extern "C" void kernel_launch(void* const* d_in, const int* in_sizes, int n_in,
                              void* d_out, int out_size, void* d_ws, size_t ws_size,
                              hipStream_t stream) {
    const float* x  = (const float*)d_in[0];
    const float* Wq = (const float*)d_in[1];
    const float* bq = (const float*)d_in[2];
    const float* Wk = (const float*)d_in[3];
    const float* bk = (const float*)d_in[4];
    const float* Wv = (const float*)d_in[5];
    const float* bv = (const float*)d_in[6];
    const float* Wo = (const float*)d_in[7];
    const float* bo = (const float*)d_in[8];

    float* outO = (float*)d_out;                        // [B,L,E]
    float* outK = outO + (size_t)BATCH * SEQ * EMB;     // [B,Hk,L,D]
    float* outV = outK + (size_t)BATCH * KVH * SEQ * HD;

    char* p = (char*)d_ws;
    __bf16* xb   = (__bf16*)p; p += (size_t)BL * EMB * 2;
    __bf16* WqbT = (__bf16*)p; p += (size_t)EMB * EMB * 2;   // [N=E][K=E]
    __bf16* WkbT = (__bf16*)p; p += (size_t)KVW * EMB * 2;   // [N=512][K=E]
    __bf16* WvbT = (__bf16*)p; p += (size_t)KVW * EMB * 2;
    __bf16* WobT = (__bf16*)p; p += (size_t)EMB * EMB * 2;
    __bf16* Qbf  = (__bf16*)p; p += (size_t)BL * EMB * 2;    // [B*L, E]
    __bf16* Kbf  = (__bf16*)p; p += (size_t)BL * KVW * 2;    // [B,Hk,L,D]
    __bf16* Vbf  = (__bf16*)p; p += (size_t)BL * KVW * 2;    // [B,Hk,D,L]
    __bf16* zbf  = (__bf16*)p; p += (size_t)BL * EMB * 2;

    // 1) conversions (weights transposed once so GEMM stages B^T row-major)
    cvt_bf16_kernel<<<2048, 256, 0, stream>>>(x, xb, (long)BL * EMB);
    cvtT_bf16_kernel<<<dim3(EMB / 32, EMB / 32), 256, 0, stream>>>(Wq, WqbT, EMB, EMB);
    cvtT_bf16_kernel<<<dim3(KVW / 32, EMB / 32), 256, 0, stream>>>(Wk, WkbT, EMB, KVW);
    cvtT_bf16_kernel<<<dim3(KVW / 32, EMB / 32), 256, 0, stream>>>(Wv, WvbT, EMB, KVW);
    cvtT_bf16_kernel<<<dim3(EMB / 32, EMB / 32), 256, 0, stream>>>(Wo, WobT, EMB, EMB);

    // 2) projections
    gemm_bf16_kernel<<<dim3(EMB / GBN, BL / GBM), 256, 0, stream>>>(
        xb, WqbT, bq, BL, EMB, EMB, nullptr, Qbf, 0, nullptr);
    gemm_bf16_kernel<<<dim3(KVW / GBN, BL / GBM), 256, 0, stream>>>(
        xb, WkbT, bk, BL, KVW, EMB, nullptr, Kbf, 1, outK);
    gemm_bf16_kernel<<<dim3(KVW / GBN, BL / GBM), 256, 0, stream>>>(
        xb, WvbT, bv, BL, KVW, EMB, nullptr, Vbf, 2, outV);

    // 3) grouped-query flash attention
    attn_kernel<<<dim3(SEQ / 128, NHEADS, BATCH), 256, 0, stream>>>(
        Qbf, Kbf, Vbf, zbf);

    // 4) output projection -> f32 d_out
    gemm_bf16_kernel<<<dim3(EMB / GBN, BL / GBM), 256, 0, stream>>>(
        zbf, WobT, bo, BL, EMB, EMB, outO, nullptr, 0, nullptr);
}
